// ATThippomod_20444044329700
// MI455X (gfx1250) — matrix operand order repaired and checked by reading, not compiled, not run on hardware
//
#include <hip/hip_runtime.h>

typedef __attribute__((ext_vector_type(16))) _Float16 v16h;
typedef __attribute__((ext_vector_type(8)))  _Float16 v8h;
typedef __attribute__((ext_vector_type(8)))  float    v8f;

#define WMMA_F16(a,b,c) __builtin_amdgcn_wmma_f32_16x16x32_f16(false,(a),false,(b),(short)0,(c),false,false)

__device__ __forceinline__ v16h hippo_make_a(const _Float16* row, int kb, int lh) {
  v8h lo = *(const v8h*)(row + kb + (lh << 4));
  v8h hi = *(const v8h*)(row + kb + 8 + (lh << 4));
  return __builtin_shufflevector(lo, hi, 0,1,2,3,4,5,6,7,8,9,10,11,12,13,14,15);
}

// ---------------------------------------------------------------------------
// Generic WMMA GEMM: C[M,N] = act(A[M,K] @ B + bias), optional f16 copy.
// A f16 row-major (lda=K, 16B-aligned rows). B pre-packed in fragment order:
// for K-tile kb and N-tile tN, halves [((kb/32)*ntn+tN)*512 + lane*16 + j].
// Fragment loads software-pipelined one K-tile ahead of the WMMA.
// ---------------------------------------------------------------------------
__global__ __launch_bounds__(128) void hippo_gemm(
    const _Float16* __restrict__ A, const _Float16* __restrict__ B,
    const float* __restrict__ bias, float* __restrict__ C, _Float16* __restrict__ Ch,
    int M, int N, int K, long sA, long sB, long sC, int act) {
  long bat = blockIdx.y;
  A += bat * sA; B += bat * sB; C += bat * sC;
  if (Ch) Ch += bat * sC;
  const int wave = threadIdx.x >> 5, lane = threadIdx.x & 31;
  const int lm = lane & 15, lh = lane >> 4;
  const int ntn = N >> 4;
  const int tile = blockIdx.x * 4 + wave;
  if (tile >= (M >> 4) * ntn) return;            // wave-uniform exit
  const int tM = tile / ntn, tN = tile % ntn;
  v8f c = {0.f,0.f,0.f,0.f,0.f,0.f,0.f,0.f};
  const _Float16* Arow = A + (size_t)(tM * 16 + lm) * K;
  const _Float16* Bt   = B + (size_t)tN * 512 + (size_t)lane * 16;
  const size_t bstep = (size_t)ntn << 9;         // halves per K-tile of packed B
  const int nk = K >> 5;
  v16h a  = hippo_make_a(Arow, 0, lh);
  v16h bf = *(const v16h*)(Bt);
  for (int kb = 1; kb < nk; ++kb) {
    __builtin_prefetch(Arow + (kb << 5) + 32, 0, 1);   // speculative; OOB dropped
    v16h a2 = hippo_make_a(Arow, kb << 5, lh);
    v16h b2 = *(const v16h*)(Bt + (size_t)kb * bstep);
    c = WMMA_F16(a, bf, c);
    a = a2; bf = b2;
  }
  c = WMMA_F16(a, bf, c);
  const float bv = bias ? bias[tN * 16 + lm] : 0.f;
  #pragma unroll
  for (int i = 0; i < 8; ++i) {
    float v = c[i] + bv;
    if (act == 1) v = fmaxf(v, 0.f);
    size_t off = (size_t)(tM * 16 + i + (lh << 3)) * N + tN * 16 + lm;
    C[off] = v;
    if (Ch) Ch[off] = (_Float16)v;
  }
}

// ---------------------------------------------------------------------------
// GRU recurrence: single workgroup, 32 waves; wave (rt,ct) owns rows rt*16..+15
// and DOUT-cols ct*16..+15; accumulates r/z/n gate tiles over K=256 with WMMA.
// h kept in LDS as f32 (state) and f16 (A fragments, b128 LDS loads).
// W_hh^T pre-packed in fragment order with N=768 (ntn=48); fragment loads
// software-pipelined one K-tile ahead of the three gate WMMAs.
// ---------------------------------------------------------------------------
__global__ __launch_bounds__(1024) void hippo_gru(
    const float* __restrict__ Gi, const _Float16* __restrict__ WhhB,
    const float* __restrict__ b_hh, const float* __restrict__ h0,
    float* __restrict__ Z) {
  __shared__ float    hf[32 * 256];
  __shared__ _Float16 hh[32 * 256];
  const int tid = threadIdx.x;
  for (int i = tid; i < 32 * 256; i += 1024) { float v = h0[i]; hf[i] = v; hh[i] = (_Float16)v; }
  __syncthreads();
  const int wave = tid >> 5, lane = tid & 31;
  const int rt = wave >> 4, ct = wave & 15;
  const int lm = lane & 15, lh = lane >> 4;
  const int d = ct * 16 + lm;
  const float bhr = b_hh[d], bhz = b_hh[256 + d], bhn = b_hh[512 + d];
  const _Float16* hrow = &hh[(rt * 16 + lm) * 256];
  const _Float16* Wb = WhhB + (size_t)lane * 16;
  for (int t = 0; t < 256; ++t) {
    v8f cr = {0.f,0.f,0.f,0.f,0.f,0.f,0.f,0.f};
    v8f cz = {0.f,0.f,0.f,0.f,0.f,0.f,0.f,0.f};
    v8f cn = {0.f,0.f,0.f,0.f,0.f,0.f,0.f,0.f};
    v16h a  = hippo_make_a(hrow, 0, lh);
    v16h br = *(const v16h*)(Wb + (size_t)(ct)      * 512);
    v16h bz = *(const v16h*)(Wb + (size_t)(16 + ct) * 512);
    v16h bn = *(const v16h*)(Wb + (size_t)(32 + ct) * 512);
    #pragma unroll
    for (int kb = 1; kb < 8; ++kb) {
      const _Float16* Wt = Wb + (size_t)kb * (48 * 512);
      v16h a2  = hippo_make_a(hrow, kb << 5, lh);
      v16h br2 = *(const v16h*)(Wt + (size_t)(ct)      * 512);
      v16h bz2 = *(const v16h*)(Wt + (size_t)(16 + ct) * 512);
      v16h bn2 = *(const v16h*)(Wt + (size_t)(32 + ct) * 512);
      cr = WMMA_F16(a, br, cr);
      cz = WMMA_F16(a, bz, cz);
      cn = WMMA_F16(a, bn, cn);
      a = a2; br = br2; bz = bz2; bn = bn2;
    }
    cr = WMMA_F16(a, br, cr);
    cz = WMMA_F16(a, bz, cz);
    cn = WMMA_F16(a, bn, cn);
    float hnew[8];
    #pragma unroll
    for (int i = 0; i < 8; ++i) {
      int b = rt * 16 + i + (lh << 3);
      const float* gi = Gi + (size_t)(t * 32 + b) * 768 + d;
      float r = 1.f / (1.f + __expf(-(gi[0]   + cr[i] + bhr)));
      float z = 1.f / (1.f + __expf(-(gi[256] + cz[i] + bhz)));
      float n = tanhf(gi[512] + r * (cn[i] + bhn));
      hnew[i] = (1.f - z) * n + z * hf[b * 256 + d];
    }
    __syncthreads();                               // all reads of hh/hf done
    #pragma unroll
    for (int i = 0; i < 8; ++i) {
      int b = rt * 16 + i + (lh << 3);
      hf[b * 256 + d] = hnew[i];
      hh[b * 256 + d] = (_Float16)hnew[i];
      Z[(size_t)b * 65536 + (size_t)t * 256 + d] = hnew[i];
    }
    __syncthreads();
  }
}

// ---------------------------------------------------------------------------
// In-place Gauss-Jordan inverse of 32 SPD 256x256 matrices (one WG per batch).
// ---------------------------------------------------------------------------
__global__ __launch_bounds__(256) void hippo_ginv(const float* __restrict__ G,
                                                  float* __restrict__ Minv) {
  __shared__ float fcol[256];
  const int b = blockIdx.x, j = threadIdx.x;
  float* M = Minv + (size_t)b * 65536;
  const float* Gb = G + (size_t)b * 65536;
  for (int i = j; i < 65536; i += 256) M[i] = Gb[i];
  __syncthreads();
  for (int k = 0; k < 256; ++k) {
    float piv = M[(size_t)k * 256 + k];
    float pv  = 1.f / piv;
    float nrkj = (j == k) ? pv : M[(size_t)k * 256 + j] * pv;
    fcol[j] = M[(size_t)j * 256 + k];
    __syncthreads();
    M[(size_t)k * 256 + j] = nrkj;
    __syncthreads();
    for (int i = 0; i < 256; ++i) {
      if (i == k) continue;
      float f = fcol[i];
      float cur = M[(size_t)i * 256 + j];
      M[(size_t)i * 256 + j] = (j == k) ? (-f * pv) : (cur - f * nrkj);
    }
    __syncthreads();
  }
}

// ---------------------------------------------------------------------------
// ODE step phase 1: time encoding, p = Zpinv @ att1 (normalized), zt, zp.
// ---------------------------------------------------------------------------
__global__ __launch_bounds__(256) void hippo_ode1(
    const float* __restrict__ times, int step, const float* __restrict__ ycur,
    const float* __restrict__ Zpinv, const float* __restrict__ Z,
    float* __restrict__ p_out, float* __restrict__ zt, _Float16* __restrict__ zt_h,
    float* __restrict__ zp) {
  __shared__ float att1[256], red[256], pl[256];
  const int b = blockIdx.x, tid = threadIdx.x;
  const float t = times[step];
  {
    float f = exp2f((float)(tid >> 1));
    float e = (tid & 1) ? cosf(f * t) : sinf(f * t);
    att1[tid] = ycur[b * 256 + tid] + e;
  }
  __syncthreads();
  const float* Zp = Zpinv + (size_t)b * 65536;
  const float* Zb = Z + (size_t)b * 65536;
  float acc = 0.f;
  { const float* row = Zp + (size_t)tid * 256;
    for (int dd = 0; dd < 256; ++dd) acc += row[dd] * att1[dd]; }
  red[tid] = acc; __syncthreads();
  for (int s = 128; s > 0; s >>= 1) { if (tid < s) red[tid] = fminf(red[tid], red[tid + s]); __syncthreads(); }
  float mn = red[0]; __syncthreads();
  float pvv = acc - mn;
  red[tid] = pvv; __syncthreads();
  for (int s = 128; s > 0; s >>= 1) { if (tid < s) red[tid] += red[tid + s]; __syncthreads(); }
  pvv = pvv / red[0];
  pl[tid] = pvv; p_out[b * 256 + tid] = pvv;
  __syncthreads();
  float a1 = 0.f, a2 = 0.f;
  for (int tt = 0; tt < 256; ++tt) {
    float pw = pl[tt];
    a1 += pw * Zp[(size_t)tt * 256 + tid];
    a2 += pw * Zb[(size_t)tt * 256 + tid];
  }
  zt[b * 256 + tid] = a1; zt_h[b * 256 + tid] = (_Float16)a1; zp[b * 256 + tid] = a2;
}

// ---------------------------------------------------------------------------
// ODE step phase 2: dy = ((Z dz) o p)^T Z - (dz.zp) zp ; Euler update.
// ---------------------------------------------------------------------------
__global__ __launch_bounds__(256) void hippo_ode2(
    const float* __restrict__ dz, const float* __restrict__ zp_in,
    const float* __restrict__ Z, const float* __restrict__ p,
    float* __restrict__ ycur, float* __restrict__ emb, int step) {
  __shared__ float dzl[256], zpl[256], wp[256], red[256];
  const int b = blockIdx.x, tid = threadIdx.x;
  dzl[tid] = dz[b * 256 + tid];
  zpl[tid] = zp_in[b * 256 + tid];
  __syncthreads();
  red[tid] = dzl[tid] * zpl[tid]; __syncthreads();
  for (int s = 128; s > 0; s >>= 1) { if (tid < s) red[tid] += red[tid + s]; __syncthreads(); }
  const float sdot = red[0];
  const float* Zb = Z + (size_t)b * 65536;
  float w = 0.f;
  { const float* row = Zb + (size_t)tid * 256;
    for (int dd = 0; dd < 256; ++dd) w += row[dd] * dzl[dd]; }
  wp[tid] = w * p[b * 256 + tid];
  __syncthreads();
  float r1 = 0.f;
  for (int tt = 0; tt < 256; ++tt) r1 += wp[tt] * Zb[(size_t)tt * 256 + tid];
  float yn = ycur[b * 256 + tid] + 0.01f * (r1 - sdot * zpl[tid]);
  ycur[b * 256 + tid] = yn;
  emb[((size_t)(step + 1) * 32 + b) * 256 + tid] = yn;
}

// ---------------------------------------------------------------------------
// Fragment-order B packers.
// Output half index: ((kb*ntn + tN)*32 + lane)*16 + j ; element (k,n) with
// k = kb*32 + j + 16*(lane>>4), n = tN*16 + (lane&15).
// ---------------------------------------------------------------------------
__global__ void hippo_packB_nk(const float* __restrict__ W, _Float16* __restrict__ Bp,
                               int N, int K, long sSrc, long sDst) {
  W += (long)blockIdx.y * sSrc; Bp += (long)blockIdx.y * sDst;
  long idx = (long)blockIdx.x * 256 + threadIdx.x;
  if (idx >= (long)N * K) return;
  int j = (int)(idx & 15);
  int lane = (int)((idx >> 4) & 31);
  long tile = idx >> 9;
  int ntn = N >> 4;
  int tN = (int)(tile % ntn);
  int kb = (int)(tile / ntn);
  int k = kb * 32 + j + ((lane >> 4) << 4);
  int n = tN * 16 + (lane & 15);
  Bp[idx] = (_Float16)W[(long)n * K + k];        // B = W^T, W is [N x K]
}

__global__ void hippo_packB_kn(const float* __restrict__ S, _Float16* __restrict__ Bp,
                               int N, int K, long sSrc, long sDst) {
  S += (long)blockIdx.y * sSrc; Bp += (long)blockIdx.y * sDst;
  long idx = (long)blockIdx.x * 256 + threadIdx.x;
  if (idx >= (long)N * K) return;
  int j = (int)(idx & 15);
  int lane = (int)((idx >> 4) & 31);
  long tile = idx >> 9;
  int ntn = N >> 4;
  int tN = (int)(tile % ntn);
  int kb = (int)(tile / ntn);
  int k = kb * 32 + j + ((lane >> 4) << 4);
  int n = tN * 16 + (lane & 15);
  Bp[idx] = (_Float16)S[(long)k * N + n];        // S is [K x N] row-major
}

// ---------------------------------------------------------------------------
// Small utility kernels
// ---------------------------------------------------------------------------
__global__ void hippo_transpose_f16(const float* __restrict__ src, _Float16* __restrict__ dst,
                                    int R, int C, long sSrc, long sDst) {
  src += (long)blockIdx.y * sSrc; dst += (long)blockIdx.y * sDst;
  long idx = (long)blockIdx.x * 256 + threadIdx.x;
  if (idx >= (long)R * C) return;
  int r = (int)(idx / C), c = (int)(idx % C);
  dst[(long)c * R + r] = (_Float16)src[idx];
}

__global__ void hippo_swap01_f16(const float* __restrict__ src, _Float16* __restrict__ dst,
                                 int A_, int B_, int D_) {
  long total = (long)A_ * B_ * D_;
  long idx = (long)blockIdx.x * 256 + threadIdx.x;
  if (idx >= total) return;
  int d = (int)(idx % D_);
  long r = idx / D_;
  int j = (int)(r % B_);
  int i = (int)(r / B_);
  dst[((long)j * A_ + i) * D_ + d] = (_Float16)src[idx];
}

__global__ void hippo_cast_f16(const float* __restrict__ s, _Float16* __restrict__ d, long n) {
  long i = (long)blockIdx.x * 256 + threadIdx.x;
  if (i < n) d[i] = (_Float16)s[i];
}

__global__ void hippo_init(const float* __restrict__ src, float* __restrict__ emb0,
                           float* __restrict__ ycur, int n) {
  int i = blockIdx.x * 256 + threadIdx.x;
  if (i < n) { float v = src[i]; emb0[i] = v; ycur[i] = v; }
}

__global__ void hippo_im2col(const _Float16* __restrict__ pred, _Float16* __restrict__ X) {
  long idx = (long)blockIdx.x * 256 + threadIdx.x;
  if (idx >= (long)1024 * 1536) return;
  int col = (int)(idx % 1536);
  long r = idx / 1536;
  int bt = (int)r;
  int b = bt >> 5, t = bt & 31;
  int i = col / 3, kk = col % 3;
  int ts = t + kk - 1;
  _Float16 v = (_Float16)0.f;
  if (ts >= 0 && ts < 32) v = pred[((long)(b * 32 + ts)) * 512 + i];
  X[idx] = v;
}

__global__ void hippo_residual(const float* __restrict__ conv, const float* __restrict__ prev,
                               float* __restrict__ outf, _Float16* __restrict__ outh, long n) {
  long i = (long)blockIdx.x * 256 + threadIdx.x;
  if (i >= n) return;
  float v = fmaxf(conv[i], 0.f) + prev[i];
  outf[i] = v; outh[i] = (_Float16)v;
}

__global__ void hippo_finalize(const float* __restrict__ ytmp, const float* __restrict__ times,
                               const float* __restrict__ emb, float* __restrict__ out) {
  long idx = (long)blockIdx.x * 256 + threadIdx.x;
  if (idx < 131072) { float v = ytmp[idx]; out[idx] = v; out[131072 + idx] = v; }
  if (idx < 32) out[262144 + idx] = times[idx];
  if (idx < 262144) {
    int d = (int)(idx & 255);
    long r = idx >> 8;
    int t = (int)(r & 31), b = (int)(r >> 5);
    out[262176 + idx] = emb[(((long)t * 32) + b) * 256 + d];
  }
}

// ---------------------------------------------------------------------------
extern "C" void kernel_launch(void* const* d_in, const int* in_sizes, int n_in,
                              void* d_out, int out_size, void* d_ws, size_t ws_size,
                              hipStream_t stream) {
  (void)in_sizes; (void)n_in; (void)out_size; (void)ws_size;
  const float* times = (const float*)d_in[0];
  const float* Y     = (const float*)d_in[1];
  const float* init_embed = (const float*)d_in[4];
  const float* h0    = (const float*)d_in[5];
  const float* W_ih  = (const float*)d_in[6];
  const float* W_hh  = (const float*)d_in[7];
  const float* b_ih  = (const float*)d_in[8];
  const float* b_hh  = (const float*)d_in[9];
  const float* Wz1 = (const float*)d_in[11]; const float* bz1 = (const float*)d_in[12];
  const float* Wz2 = (const float*)d_in[13]; const float* bz2 = (const float*)d_in[14];
  const float* Wz3 = (const float*)d_in[15]; const float* bz3 = (const float*)d_in[16];
  const float* Wo1 = (const float*)d_in[17]; const float* bo1 = (const float*)d_in[18];
  const float* Wc1 = (const float*)d_in[19]; const float* bc1 = (const float*)d_in[20];
  const float* Wc2 = (const float*)d_in[21]; const float* bc2 = (const float*)d_in[22];
  const float* Wo3 = (const float*)d_in[23]; const float* bo3 = (const float*)d_in[24];
  float* out = (float*)d_out;

  size_t off = 0;
  auto alloc = [&](size_t nbytes) -> char* {
    off = (off + 255) & ~(size_t)255;
    char* p = (char*)d_ws + off; off += nbytes; return p;
  };
  // f16 packed weights / activations
  _Float16* Yh    = (_Float16*)alloc(1048576 * 2);
  _Float16* WihB  = (_Float16*)alloc(98304 * 2);
  _Float16* WhhB  = (_Float16*)alloc(196608 * 2);
  _Float16* Wz1B  = (_Float16*)alloc(131072 * 2);
  _Float16* Wz2B  = (_Float16*)alloc(262144 * 2);
  _Float16* Wz3B  = (_Float16*)alloc(131072 * 2);
  _Float16* Wo1B  = (_Float16*)alloc(131072 * 2);
  _Float16* Wo3B  = (_Float16*)alloc(65536 * 2);
  _Float16* Wc1B  = (_Float16*)alloc(786432 * 2);
  _Float16* Wc2B  = (_Float16*)alloc(786432 * 2);
  // f32 big buffers
  float* Gi    = (float*)alloc(6291456 * 4);      // reused for output head below
  float* Z     = (float*)alloc(2097152 * 4);
  _Float16* Zh   = (_Float16*)alloc(2097152 * 2); // row-major (pinv A-side)
  _Float16* ZgB  = (_Float16*)alloc(2097152 * 2); // fragment-packed (gram B-side)
  _Float16* ZhT  = (_Float16*)alloc(2097152 * 2); // row-major Z^T (gram A-side)
  float* G     = (float*)alloc(2097152 * 4);
  float* Ginv  = (float*)alloc(2097152 * 4);
  _Float16* GinvB = (_Float16*)alloc(2097152 * 2); // fragment-packed (pinv B-side)
  float* Zpinv = (float*)alloc(2097152 * 4);
  // small ODE buffers
  float* pbuf = (float*)alloc(8192 * 4);
  float* zt   = (float*)alloc(8192 * 4);
  _Float16* zth = (_Float16*)alloc(8192 * 2);
  float* zp   = (float*)alloc(8192 * 4);
  float* h1f  = (float*)alloc(16384 * 4);
  _Float16* h1h = (_Float16*)alloc(16384 * 2);
  float* h2f  = (float*)alloc(16384 * 4);
  _Float16* h2h = (_Float16*)alloc(16384 * 2);
  float* dzf  = (float*)alloc(8192 * 4);
  float* ycur = (float*)alloc(8192 * 4);
  float* emb  = (float*)alloc(262144 * 4);
  // output-head buffers carved from Gi region (Gi dead after GRU)
  size_t hoff = 0;
  auto halloc = [&](size_t nbytes) -> char* {
    hoff = (hoff + 255) & ~(size_t)255;
    char* p = (char*)Gi + hoff; hoff += nbytes; return p;
  };
  _Float16* Eh    = (_Float16*)halloc(262144 * 2);
  float* predA    = (float*)halloc(524288 * 4);
  _Float16* predAh = (_Float16*)halloc(524288 * 2);
  float* predB    = (float*)halloc(524288 * 4);
  _Float16* predBh = (_Float16*)halloc(524288 * 2);
  float* predC    = (float*)halloc(524288 * 4);
  _Float16* predCh = (_Float16*)halloc(524288 * 2);
  float* convt    = (float*)halloc(524288 * 4);
  _Float16* Xcol  = (_Float16*)halloc(1572864 * 2);
  float* ytmp     = (float*)halloc(131072 * 4);

  auto gemm = [&](const _Float16* A, const _Float16* B, const float* bias,
                  float* C, _Float16* Ch, int M, int N, int K,
                  long sA, long sB, long sC, int bat, int act) {
    int tiles = (M / 16) * (N / 16);
    dim3 g((tiles + 3) / 4, bat);
    hippo_gemm<<<g, 128, 0, stream>>>(A, B, bias, C, Ch, M, N, K, sA, sB, sC, act);
  };
  auto packNK = [&](const float* W, _Float16* Bp, int N, int K) {
    dim3 g(((long)N * K + 255) / 256, 1);
    hippo_packB_nk<<<g, 256, 0, stream>>>(W, Bp, N, K, 0, 0);
  };

  // ---- weight packing (fragment order) ----
  hippo_swap01_f16<<<4096, 256, 0, stream>>>(Y, Yh, 32, 256, 128);   // (B,T,DIN)->(T,B,DIN)
  packNK(W_ih, WihB, 768, 128);
  packNK(W_hh, WhhB, 768, 256);
  packNK(Wz1, Wz1B, 512, 256);
  packNK(Wz2, Wz2B, 512, 512);
  packNK(Wz3, Wz3B, 256, 512);
  packNK(Wo1, Wo1B, 512, 256);
  packNK(Wo3, Wo3B, 128, 512);
  packNK(Wc1, Wc1B, 512, 1536);
  packNK(Wc2, Wc2B, 512, 1536);

  // ---- GRU: hoisted input projection then sequential recurrence ----
  gemm(Yh, WihB, b_ih, Gi, nullptr, 8192, 768, 128, 0, 0, 0, 1, 0);
  hippo_gru<<<1, 1024, 0, stream>>>(Gi, WhhB, b_hh, h0, Z);

  // ---- pinv via normal equations: Zpinv = Z (Z^T Z)^-1  (Wv == ones) ----
  hippo_cast_f16<<<8192, 256, 0, stream>>>(Z, Zh, 2097152);
  { dim3 g(256, 32); hippo_packB_kn<<<g, 256, 0, stream>>>(Z, ZgB, 256, 256, 65536, 65536); }
  { dim3 g(256, 32); hippo_transpose_f16<<<g, 256, 0, stream>>>(Z, ZhT, 256, 256, 65536, 65536); }
  gemm(ZhT, ZgB, nullptr, G, nullptr, 256, 256, 256, 65536, 65536, 65536, 32, 0);
  hippo_ginv<<<32, 256, 0, stream>>>(G, Ginv);
  { dim3 g(256, 32); hippo_packB_kn<<<g, 256, 0, stream>>>(Ginv, GinvB, 256, 256, 65536, 65536); }
  gemm(Zh, GinvB, nullptr, Zpinv, nullptr, 256, 256, 256, 65536, 65536, 65536, 32, 0);

  // ---- Euler ODE (Jacobian contraction folded algebraically) ----
  hippo_init<<<32, 256, 0, stream>>>(init_embed, emb, ycur, 8192);
  for (int s = 0; s < 31; ++s) {
    hippo_ode1<<<32, 256, 0, stream>>>(times, s, ycur, Zpinv, Z, pbuf, zt, zth, zp);
    gemm(zth, Wz1B, bz1, h1f, h1h, 32, 512, 256, 0, 0, 0, 1, 1);
    gemm(h1h, Wz2B, bz2, h2f, h2h, 32, 512, 512, 0, 0, 0, 1, 1);
    gemm(h2h, Wz3B, bz3, dzf, nullptr, 32, 256, 512, 0, 0, 0, 1, 0);
    hippo_ode2<<<32, 256, 0, stream>>>(dzf, zp, Z, pbuf, ycur, emb, s);
  }

  // ---- output head ----
  hippo_swap01_f16<<<1024, 256, 0, stream>>>(emb, Eh, 32, 32, 256);  // (T,B,D)->(B,T,D)
  gemm(Eh, Wo1B, bo1, predA, predAh, 1024, 512, 256, 0, 0, 0, 1, 1);
  hippo_im2col<<<6144, 256, 0, stream>>>(predAh, Xcol);
  gemm(Xcol, Wc1B, bc1, convt, nullptr, 1024, 512, 1536, 0, 0, 0, 1, 0);
  hippo_residual<<<2048, 256, 0, stream>>>(convt, predA, predB, predBh, 524288);
  hippo_im2col<<<6144, 256, 0, stream>>>(predBh, Xcol);
  gemm(Xcol, Wc2B, bc2, convt, nullptr, 1024, 512, 1536, 0, 0, 0, 1, 0);
  hippo_residual<<<2048, 256, 0, stream>>>(convt, predB, predC, predCh, 524288);
  gemm(predCh, Wo3B, bo3, ytmp, nullptr, 1024, 128, 512, 0, 0, 0, 1, 0);

  hippo_finalize<<<1024, 256, 0, stream>>>(ytmp, times, emb, out);
}
